// CrossLayerRouter_64141041598877
// MI455X (gfx1250) — compile-verified, matrix operand order checked by reading, not run on hardware
//
#include <hip/hip_runtime.h>
#include <stdint.h>

// ---------------------------------------------------------------------------
// MoE noisy-top-k router for MI455X (gfx1250, wave32, WMMA).
//
// GEMM path: fp32 x is split into bf16 hi/lo on the fly; weights are
// pre-packed (kernel 1) into bf16 hi/lo V_WMMA B-fragments in d_ws.
// Product emulated as hi*hi + hi*lo + lo*hi with v_wmma_f32_16x16x32_bf16
// (f32 accumulate), ~2^-16 relative accuracy at ~2.7x the rate of native
// v_wmma_f32_16x16x4_f32 -> near the 23.3 TB/s HBM roofline (~12 us).
//
// Data movement: double-buffered GLOBAL_LOAD_ASYNC_TO_LDS_B128 (ASYNCcnt)
// overlaps the HBM fetch of chunk k+1 with the WMMA compute of chunk k.
// Falls back to synchronous loads if the async builtins are unavailable.
// ---------------------------------------------------------------------------

#define BDIM   8
#define SDIM   4096
#define DDIM   2048
#define EDIM   64
#define KTOP   8
#define NTOK   (BDIM * SDIM)        // 32768 tokens
#define TILE_M 64                   // tokens per workgroup (4 waves x 16)
#define KCHUNK 32                   // K per bf16 WMMA
#define NCHUNK (DDIM / KCHUNK)      // 64 chunks
#define NT     8                    // 8 column tiles of 16 -> 128 cols (Wr|Wn)
#define XROW   36                   // padded LDS row stride (16B-aligned rows)
#define WPK_DW_PER_CHUNK (NT * 2 * 32 * 8)  // 4096 dwords = 16 KB / chunk

typedef __attribute__((ext_vector_type(16))) __bf16 v16bf;
typedef __attribute__((ext_vector_type(8)))  float  v8f;

#if __has_builtin(__builtin_amdgcn_global_load_async_to_lds_b128) && \
    __has_builtin(__builtin_amdgcn_s_wait_asynccnt)
#define ASYNC_PATH 1
// Probe-derived signature: (int4 AS1* src, int4 AS3* dst, imm offset, imm cpol)
typedef int v4i __attribute__((ext_vector_type(4)));
typedef __attribute__((address_space(1))) v4i gas_v4i;
typedef __attribute__((address_space(3))) v4i las_v4i;
__device__ __forceinline__ void cp16_async(void* dst_lds, const void* src_g) {
  __builtin_amdgcn_global_load_async_to_lds_b128(
      (gas_v4i*)src_g, (las_v4i*)dst_lds, 0, 0);
}
#else
#define ASYNC_PATH 0
#endif

// ---------------------------------------------------------------------------
// Kernel 1: pack [Wr | Wn] (D x 128, fp32) into bf16 hi/lo WMMA B-fragments.
// Layout: wpk[kb][nt][part][lane][p]  (p = packed dword = 2 bf16).
// Fragment mapping (mirror of ISA 16-bit A layout, lane = column):
//   col = nt*16 + (lane&15), h = lane>>4,
//   dword p holds K = (p/4)*16 + h*8 + (p%4)*2  and K+1.
// ---------------------------------------------------------------------------
__global__ __launch_bounds__(256) void pack_w_kernel(
    const float* __restrict__ Wr, const float* __restrict__ Wn,
    uint32_t* __restrict__ wpk)
{
  int t = blockIdx.x * 256 + threadIdx.x;
  if (t >= NCHUNK * WPK_DW_PER_CHUNK) return;
  const int p    = t & 7;
  const int lane = (t >> 3) & 31;
  const int part = (t >> 8) & 1;
  const int nt   = (t >> 9) & 7;
  const int kb   = t >> 12;

  const int col = nt * 16 + (lane & 15);        // 0..127
  const int h   = lane >> 4;
  const int kl  = ((p >> 2) * 16) + h * 8 + (p & 3) * 2;
  const int d0  = kb * KCHUNK + kl;

  const float* src = (col < EDIM) ? Wr : Wn;
  const int c = col & (EDIM - 1);
  const float f0 = src[(size_t)d0 * EDIM + c];
  const float f1 = src[(size_t)(d0 + 1) * EDIM + c];

  const __bf16 h0 = (__bf16)f0, h1 = (__bf16)f1;
  __bf16 o0, o1;
  if (part == 0) { o0 = h0; o1 = h1; }
  else { o0 = (__bf16)(f0 - (float)h0); o1 = (__bf16)(f1 - (float)h1); }

  const uint32_t u0 = (uint32_t)__builtin_bit_cast(uint16_t, o0);
  const uint32_t u1 = (uint32_t)__builtin_bit_cast(uint16_t, o1);
  wpk[t] = u0 | (u1 << 16);
}

// ---------------------------------------------------------------------------
// Kernel 2: main router.
// ---------------------------------------------------------------------------
__global__ __launch_bounds__(128) void router_kernel(
    const float* __restrict__ x,   const float* __restrict__ eps,
    const float* __restrict__ br,  const float* __restrict__ bn,
    const float* __restrict__ Ws,  const float* __restrict__ bs,
    const uint32_t* __restrict__ wpk, float* __restrict__ out)
{
  __shared__ __align__(16) union {
    struct {
      float    xbuf[2][TILE_M * XROW];        // 2 x 9216 B  fp32 x tiles
      uint32_t wbuf[2][WPK_DW_PER_CHUNK];     // 2 x 16 KB   packed bf16 W
      float    wsall[DDIM];                   // 8 KB        full Ws vector
    } lb;
    float noisy[TILE_M * 2 * EDIM];           // 32 KB  logits|noise -> noisy
  } sm;
  __shared__ float skipred[128];

  const int tid   = threadIdx.x;
  const int lane  = tid & 31;
  const int w     = tid >> 5;             // wave id 0..3
  const int h     = lane >> 4;            // half-wave
  const int m     = lane & 15;            // row within M-tile / column in tile
  const int tile0 = blockIdx.x * TILE_M;

  v8f acc[NT] = {};                       // 8 x (16x16 f32) accumulators
  float sacc = 0.f;                       // skip-logit partial (half token)

#if ASYNC_PATH
  // ---- one-time Ws preload + chunk-0 async issue ------------------------
  #pragma unroll
  for (int it = 0; it < 4; ++it) {
    const int idx = (it * 128 + tid) * 4;                 // float index
    cp16_async(&sm.lb.wsall[idx], Ws + idx);
  }
  #pragma unroll
  for (int it = 0; it < 4; ++it) {
    const int idx = it * 128 + tid;                       // 0..511
    const int row = idx >> 3;
    const int c4  = (idx & 7) * 4;
    cp16_async(&sm.lb.xbuf[0][row * XROW + c4],
               x + (size_t)(tile0 + row) * DDIM + c4);
  }
  #pragma unroll
  for (int it = 0; it < 8; ++it) {
    const int idx = (it * 128 + tid) * 4;                 // dword index
    cp16_async(&sm.lb.wbuf[0][idx], wpk + idx);
  }
#else
  for (int i = tid; i < DDIM; i += 128) sm.lb.wsall[i] = Ws[i];
#endif

  for (int kb = 0; kb < NCHUNK; ++kb) {
    const int cb = kb & 1;

#if ASYNC_PATH
    // ---- issue async loads for chunk kb+1, then wait for chunk kb -------
    if (kb + 1 < NCHUNK) {
      const int nb = (kb + 1) & 1;
      #pragma unroll
      for (int it = 0; it < 4; ++it) {
        const int idx = it * 128 + tid;
        const int row = idx >> 3;
        const int c4  = (idx & 7) * 4;
        cp16_async(&sm.lb.xbuf[nb][row * XROW + c4],
                   x + (size_t)(tile0 + row) * DDIM + (kb + 1) * KCHUNK + c4);
      }
      const uint32_t* wsrc = wpk + (size_t)(kb + 1) * WPK_DW_PER_CHUNK;
      #pragma unroll
      for (int it = 0; it < 8; ++it) {
        const int idx = (it * 128 + tid) * 4;
        cp16_async(&sm.lb.wbuf[nb][idx], wsrc + idx);
      }
      __builtin_amdgcn_s_wait_asynccnt(12);   // 12 newer in flight, kb done
    } else {
      __builtin_amdgcn_s_wait_asynccnt(0);
    }
    __syncthreads();                          // publish chunk kb to all waves
#else
    __syncthreads();                          // prev compute done reading LDS
    #pragma unroll
    for (int it = 0; it < 4; ++it) {
      const int idx = it * 128 + tid;
      const int row = idx >> 3;
      const int c4  = (idx & 7) * 4;
      const float4 v = *(const float4*)(x + (size_t)(tile0 + row) * DDIM
                                          + kb * KCHUNK + c4);
      float* dst = &sm.lb.xbuf[cb][row * XROW + c4];
      dst[0] = v.x; dst[1] = v.y; dst[2] = v.z; dst[3] = v.w;
    }
    {
      const uint4* src = (const uint4*)(wpk + (size_t)kb * WPK_DW_PER_CHUNK);
      uint4* dst = (uint4*)sm.lb.wbuf[cb];
      #pragma unroll
      for (int it = 0; it < 8; ++it) dst[it * 128 + tid] = src[it * 128 + tid];
    }
    if (kb + 1 < NCHUNK) {
      const float* np = x + (size_t)(tile0 + (tid >> 1)) * DDIM
                          + (kb + 1) * KCHUNK + (tid & 1) * 16;
      __builtin_prefetch(np, 0, 0);
    }
    __syncthreads();
#endif

    // ---- build A fragments (hi/lo bf16 split of this wave's 16 rows) ----
    v16bf ahi, alo;
    const float* xrow = &sm.lb.xbuf[cb][(w * 16 + m) * XROW];
    #pragma unroll
    for (int p = 0; p < 8; ++p) {
      const int k = ((p >> 2) * 16) + h * 8 + (p & 3) * 2;
      const float f0 = xrow[k], f1 = xrow[k + 1];
      const __bf16 b0 = (__bf16)f0, b1 = (__bf16)f1;
      ahi[2 * p]     = b0;
      ahi[2 * p + 1] = b1;
      alo[2 * p]     = (__bf16)(f0 - (float)b0);
      alo[2 * p + 1] = (__bf16)(f1 - (float)b1);
    }

    // ---- skip projection rides on the same x tile -----------------------
    #pragma unroll
    for (int k = 0; k < 16; ++k)
      sacc += xrow[h * 16 + k] * sm.lb.wsall[kb * KCHUNK + h * 16 + k];

    // ---- 8 column tiles x 3 WMMAs (hi*hi + hi*lo + lo*hi) --------------
    #pragma unroll
    for (int nt = 0; nt < NT; ++nt) {
      union { uint32_t u[8]; v16bf v; } Bh, Bl;
      const uint32_t* ph = &sm.lb.wbuf[cb][((nt * 2 + 0) * 32 + lane) * 8];
      const uint32_t* pl = &sm.lb.wbuf[cb][((nt * 2 + 1) * 32 + lane) * 8];
      #pragma unroll
      for (int q = 0; q < 8; ++q) { Bh.u[q] = ph[q]; Bl.u[q] = pl[q]; }
      acc[nt] = __builtin_amdgcn_wmma_f32_16x16x32_bf16(
          false, ahi, false, Bh.v, (short)0, acc[nt], false, false);
      acc[nt] = __builtin_amdgcn_wmma_f32_16x16x32_bf16(
          false, ahi, false, Bl.v, (short)0, acc[nt], false, false);
      acc[nt] = __builtin_amdgcn_wmma_f32_16x16x32_bf16(
          false, alo, false, Bh.v, (short)0, acc[nt], false, false);
    }

#if ASYNC_PATH
    __syncthreads();   // all waves done with buf cb before iter kb+1 refills
#endif
  }

  skipred[tid] = sacc;
  __syncthreads();                        // all waves done reading loop LDS

  // ---- dump C fragments: VGPR r -> M = r + 8h, N = nt*16 + m ------------
  #pragma unroll
  for (int nt = 0; nt < NT; ++nt)
    #pragma unroll
    for (int r = 0; r < 8; ++r)
      sm.noisy[(w * 16 + r + 8 * h) * (2 * EDIM) + nt * 16 + m] = acc[nt][r];
  __syncthreads();

  // ---- epilogue: one lane per token -------------------------------------
  if (lane < 16) {
    const int row = w * 16 + lane;
    const int tok = tile0 + row;
    float* nz = &sm.noisy[row * (2 * EDIM)];
    const float* ep = eps + (size_t)tok * EDIM;

    // noisy = (xWr + br) + eps * softplus(xWn + bn), in place
    for (int e = 0; e < EDIM; ++e) {
      const float lg = nz[e] + br[e];
      const float nv = nz[EDIM + e] + bn[e];
      const float sp = (nv > 0.f) ? (nv + log1pf(__expf(-nv)))
                                  : log1pf(__expf(nv));
      nz[e] = lg + ep[e] * sp;
    }

    // destructive 8-pass argmax (matches lax.top_k ordering / tie rule)
    float val[KTOP]; int idx[KTOP];
    #pragma unroll
    for (int kk = 0; kk < KTOP; ++kk) {
      float best = -__builtin_inff(); int bi = 0;
      for (int e = 0; e < EDIM; ++e) {
        const float v = nz[e];
        if (v > best) { best = v; bi = e; }
      }
      val[kk] = best; idx[kk] = bi;
      nz[bi] = -__builtin_inff();
    }

    // sparse softmax over the top-8
    const float mx = val[0];
    float s = 0.f;
    #pragma unroll
    for (int kk = 0; kk < KTOP; ++kk) s += __expf(val[kk] - mx);
    const float inv = 1.f / s;

    float* orow = out + (size_t)tok * EDIM;
    const float4 z = {0.f, 0.f, 0.f, 0.f};
    #pragma unroll
    for (int q = 0; q < EDIM / 4; ++q) ((float4*)orow)[q] = z;
    float* irow = out + (size_t)NTOK * EDIM + (size_t)tok * KTOP;
    #pragma unroll
    for (int kk = 0; kk < KTOP; ++kk) {
      orow[idx[kk]] = __expf(val[kk] - mx) * inv;
      irow[kk] = (float)idx[kk];
    }

    // skip_prob = sigmoid(x.Ws + bs); combine the two half-token sums
    const float sl = skipred[w * 32 + lane] + skipred[w * 32 + lane + 16] + bs[0];
    out[(size_t)NTOK * (EDIM + KTOP) + tok] = 1.f / (1.f + __expf(-sl));
  }
}

// ---------------------------------------------------------------------------
extern "C" void kernel_launch(void* const* d_in, const int* in_sizes, int n_in,
                              void* d_out, int out_size, void* d_ws, size_t ws_size,
                              hipStream_t stream) {
  const float* x   = (const float*)d_in[0];
  const float* eps = (const float*)d_in[1];
  const float* Wr  = (const float*)d_in[2];
  const float* br  = (const float*)d_in[3];
  const float* Wn  = (const float*)d_in[4];
  const float* bn  = (const float*)d_in[5];
  const float* Ws  = (const float*)d_in[6];
  const float* bs  = (const float*)d_in[7];

  uint32_t* wpk = (uint32_t*)d_ws;        // 1 MB packed weights

  const int pack_total = NCHUNK * WPK_DW_PER_CHUNK;     // 262144
  pack_w_kernel<<<pack_total / 256, 256, 0, stream>>>(Wr, Wn, wpk);

  router_kernel<<<NTOK / TILE_M, 128, 0, stream>>>(
      x, eps, br, bn, Ws, bs, wpk, (float*)d_out);
}
